// InterlacedPoolAttention_31748398252238
// MI455X (gfx1250) — matrix operand order repaired and checked by reading, not compiled, not run on hardware
//
#include <hip/hip_runtime.h>
#include <hip/hip_fp16.h>

// ---------------- problem constants (shapes fixed by setup_inputs) ------------
#define B_    8
#define HH_   224
#define WW_   224
#define C_    128
#define NPIX  (HH_*WW_)          // 50176
#define NTOK  (B_*NPIX)          // 401408
#define WS_   7
#define NH_   4
#define NWH   32                 // windows per spatial side (224/7)
#define NWIN  (B_*NWH*NWH)       // 8192
#define LW    (WS_*WS_)          // 49 tokens / window (padded to 64 on chip)

// ---------------- workspace layout (bytes, 256-aligned) -----------------------
#define OFF_POOLED 0u
#define OFF_SCA    4096u
#define OFF_SAIN   8192u                          // 8*2*50176*4 = 3211264
#define OFF_SASIG  (OFF_SAIN + 3211264u)          // 8*50176*4   = 1605632
#define OFF_WIN    (OFF_SASIG + 1605632u)         // 384*128*2   = 98304
#define OFF_WOUT   (OFF_WIN + 98304u)             // 128*128*2   = 32768
#define OFF_XW     (OFF_WOUT + 32768u)            // 401408*128*2 = 102760448
// total ws usage ~107.7 MB

// ---------------- WMMA plumbing ----------------------------------------------
typedef __attribute__((ext_vector_type(16))) _Float16 v16h;
typedef __attribute__((ext_vector_type(8)))  float    v8f;

union FragH { v16h v; uint32_t u[8]; };

__device__ __forceinline__ v8f zero8() {
  v8f z = {0.f,0.f,0.f,0.f,0.f,0.f,0.f,0.f};
  return z;
}
__device__ __forceinline__ v8f wmma16(v16h a, v16h b, v8f c) {
  // D = A(16x32 f16) * B(32x16 f16) + C(16x16 f32)
  return __builtin_amdgcn_wmma_f32_16x16x32_f16(false, a, false, b, (short)0, c,
                                                false, false);
}

// A fragment, 16x32 f16, from row-major halfs (stride in halfs, even).
// lanes 0-15: M=lane, K 0..15; lanes 16-31: M=lane-16, K 16..31
__device__ __forceinline__ v16h load_a_lds(const _Float16* base, int row0, int k0,
                                           int stride) {
  int lane = threadIdx.x & 31;
  const _Float16* rp = base + (row0 + (lane & 15)) * stride + k0 + ((lane >> 4) << 4);
  FragH f;
#pragma unroll
  for (int j = 0; j < 8; ++j) {
    int k = j << 1;   // j -> 2j
    f.u[j] = *(const uint32_t*)(rp + k);
  }
  return f.v;
}

// B fragment, 32x16 f16, from row-major (N,K) memory (K contiguous).
// lanes 0-15: N=lane, K=0..15 ; lanes 16-31: N=lane-16, K=16..31 (VGPR j -> K=2j,2j+1)
__device__ __forceinline__ v16h load_b_rowmajor(const _Float16* W, int n0, int k0,
                                                int ldk) {
  int lane = threadIdx.x & 31;
  const _Float16* rp = W + (n0 + (lane & 15)) * ldk + k0 + ((lane >> 4) << 4);
  FragH f;
#pragma unroll
  for (int j = 0; j < 8; ++j) f.u[j] = *(const uint32_t*)(rp + (j << 1));
  return f.v;
}

// B fragment for S = Q*K^T : K-dim = d (32), N = key token.
// kp layout (per head): element (d,l) at (d>>1)*128 + l*2 + (d&1)  -> d-pairs contiguous
__device__ __forceinline__ v16h load_b_kp(const _Float16* kp, int n0) {
  int lane = threadIdx.x & 31;
  int l = n0 + (lane & 15);
  int kb2 = (lane >> 4) << 3;                 // d/2 base (0 or 8)
  FragH f;
#pragma unroll
  for (int j = 0; j < 8; ++j) f.u[j] = *(const uint32_t*)(kp + (kb2 + j)*128 + (l << 1));
  return f.v;
}

// B fragment for O = P*V : K-dim = key token (64), N = d.
// vp layout (per head): element (l,d) at (l>>1)*64 + d*2 + (l&1) -> token-pairs contiguous
__device__ __forceinline__ v16h load_b_vp(const _Float16* vp, int n0, int k0) {
  int lane = threadIdx.x & 31;
  int d = n0 + (lane & 15);
  int kb2 = (k0 >> 1) + ((lane >> 4) << 3);   // l/2 base
  FragH f;
#pragma unroll
  for (int j = 0; j < 8; ++j) f.u[j] = *(const uint32_t*)(vp + (kb2 + j)*64 + (d << 1));
  return f.v;
}

// ---------------- prep kernels ------------------------------------------------
__global__ void k_zero(float* __restrict__ p, int n) {
  int i = blockIdx.x * 256 + threadIdx.x;
  if (i < n) p[i] = 0.f;
}

// channel mean-pool partial sums: grid (64 segs, B), 128 threads = channels
__global__ void k_pool(const float* __restrict__ x, float* __restrict__ pooled) {
  int b = blockIdx.y, seg = blockIdx.x, c = threadIdx.x;
  const float* xp = x + ((size_t)b * NPIX + (size_t)seg * 784) * C_ + c;
  float s = 0.f;
  for (int r = 0; r < 784; ++r) s += xp[(size_t)r * C_];
  atomicAdd(&pooled[b * C_ + c], s);
}

// channel-attention MLP: sigmoid(relu(p@W1^T+b1)@W2^T+b2)
__global__ void k_ca(const float* __restrict__ pooled, const float* __restrict__ w1,
                     const float* __restrict__ b1, const float* __restrict__ w2,
                     const float* __restrict__ b2, float* __restrict__ sca) {
  __shared__ float pl[C_];
  __shared__ float hb[16];
  int b = blockIdx.x, tid = threadIdx.x;
  pl[tid] = pooled[b * C_ + tid] * (1.f / (float)NPIX);
  __syncthreads();
  if (tid < 16) {
    float h = b1[tid];
    for (int c = 0; c < C_; ++c) h += pl[c] * w1[tid * C_ + c];
    hb[tid] = fmaxf(h, 0.f);
  }
  __syncthreads();
  float s = b2[tid];
  for (int j = 0; j < 16; ++j) s += hb[j] * w2[tid * 16 + j];
  sca[b * C_ + tid] = 1.f / (1.f + __expf(-s));
}

// per-pixel channel mean/max of the CA-scaled tensor (spatial-attn input maps)
__global__ void k_stats(const float* __restrict__ x, const float* __restrict__ sca,
                        float* __restrict__ sain) {
  __shared__ float sc[C_];
  int pix = blockIdx.x * 256 + threadIdx.x;          // 50176 % 256 == 0 -> uniform b
  int b = pix / NPIX;
  if (threadIdx.x < C_) sc[threadIdx.x] = sca[b * C_ + threadIdx.x];
  __syncthreads();
  int n = pix - b * NPIX;
  const float4* xv = (const float4*)x + (size_t)pix * 32;
  float sum = 0.f, mx = -3.4e38f;
  for (int c4 = 0; c4 < 32; ++c4) {
    float4 v = xv[c4];
    float s0 = v.x * sc[c4*4+0], s1 = v.y * sc[c4*4+1];
    float s2 = v.z * sc[c4*4+2], s3 = v.w * sc[c4*4+3];
    sum += s0 + s1 + s2 + s3;
    mx = fmaxf(mx, fmaxf(fmaxf(s0, s1), fmaxf(s2, s3)));
  }
  sain[(size_t)b * 2 * NPIX + n]        = sum * (1.f / C_);
  sain[(size_t)b * 2 * NPIX + NPIX + n] = mx;
}

// 7x7 conv (2 in ch -> 1), SAME pad 3, + bias, sigmoid
__global__ void k_conv(const float* __restrict__ sain, const float* __restrict__ saw,
                       const float* __restrict__ sab, float* __restrict__ sasig) {
  __shared__ float wsh[98];
  if (threadIdx.x < 98) wsh[threadIdx.x] = saw[threadIdx.x];
  __syncthreads();
  int pix = blockIdx.x * 256 + threadIdx.x;
  int b = pix / NPIX;
  int n = pix - b * NPIX;
  int y = n / WW_, xx = n - y * WW_;
  float acc = sab[0];
  for (int ic = 0; ic < 2; ++ic) {
    const float* plane = sain + (size_t)b * 2 * NPIX + (size_t)ic * NPIX;
    for (int ky = 0; ky < 7; ++ky) {
      int yy = y + ky - 3;
      if (yy < 0 || yy >= HH_) continue;
      for (int kx = 0; kx < 7; ++kx) {
        int xq = xx + kx - 3;
        if (xq < 0 || xq >= WW_) continue;
        acc += plane[yy * WW_ + xq] * wsh[ic * 49 + ky * 7 + kx];
      }
    }
  }
  sasig[pix] = 1.f / (1.f + __expf(-acc));
}

// f32 -> f16 weight conversion (in_proj 384x128 and out_proj 128x128, row-major)
__global__ void k_wprep(const float* __restrict__ w1, const float* __restrict__ w2,
                        _Float16* __restrict__ h1, _Float16* __restrict__ h2) {
  int i = blockIdx.x * 256 + threadIdx.x;
  if (i < 384 * C_) h1[i] = (_Float16)w1[i];
  if (i < C_ * C_)  h2[i] = (_Float16)w2[i];
}

// apply both gates, window-partition, f16 pack: xw[win][l][c]
__global__ void k_pack(const float* __restrict__ x, const float* __restrict__ sca,
                       const float* __restrict__ sasig, _Float16* __restrict__ xw) {
  int g = blockIdx.x * 256 + threadIdx.x;     // NTOK*32 float4-lanes
  int t = g >> 5, c4 = g & 31;
  int win = t / LW, l = t - win * LW;
  int b = win >> 10;
  int hh = (win >> 5) & 31, ww = win & 31;
  int li = l / WS_, lj = l - li * WS_;
  int n = (hh * WS_ + li) * WW_ + (ww * WS_ + lj);
  int pix = b * NPIX + n;
  float4 v = ((const float4*)x)[(size_t)pix * 32 + c4];
  float g2 = sasig[pix];
  float s0 = sca[b * C_ + c4*4+0] * g2, s1 = sca[b * C_ + c4*4+1] * g2;
  float s2 = sca[b * C_ + c4*4+2] * g2, s3 = sca[b * C_ + c4*4+3] * g2;
  union { _Float16 h[4]; uint2 u; } o;
  o.h[0] = (_Float16)(v.x * s0); o.h[1] = (_Float16)(v.y * s1);
  o.h[2] = (_Float16)(v.z * s2); o.h[3] = (_Float16)(v.w * s3);
  ((uint2*)xw)[(size_t)t * 32 + c4] = o.u;
}

// ---------------- fused window attention -------------------------------------
#define LDP  136   // padded half-stride for 128-col LDS tiles (bank-conflict free)
#define PSTR 72    // padded half-stride for 64-col P tiles
#define SMEM_BYTES (((64*LDP*3) + (4*2048*2) + (4*64*PSTR)) * 2 + 676 * 4)  // 124560

__global__ void __launch_bounds__(256)
win_attn(const _Float16* __restrict__ xw,
         const _Float16* __restrict__ Wqkv, const float* __restrict__ bqkv,
         const _Float16* __restrict__ Wout, const float* __restrict__ bout,
         const float* __restrict__ rpe, float* __restrict__ out) {
  extern __shared__ char smemraw[];
  _Float16* xw_lds = (_Float16*)smemraw;          // 64 x LDP  (window tokens)
  _Float16* q_lds  = xw_lds + 64 * LDP;           // 64 x LDP  (pre-scaled Q)
  _Float16* kp_lds = q_lds  + 64 * LDP;           // 4 heads x 2048 (d-pair interleaved)
  _Float16* vp_lds = kp_lds + 4 * 2048;           // 4 heads x 2048 (tok-pair interleaved)
  _Float16* p_lds  = vp_lds + 4 * 2048;           // 4 heads x 64 x PSTR (softmax P)
  _Float16* ao_lds = p_lds  + 4 * 64 * PSTR;      // 64 x LDP  (attention output)
  float*    rpe_lds = (float*)(ao_lds + 64 * LDP);// 169*4 bias table

  const int tid = threadIdx.x;
  const int lane = tid & 31;
  const int wv  = tid >> 5;                       // 8 wave32s
  const int win = blockIdx.x;

  __builtin_prefetch(Wout + ((tid & 127) << 7), 0, 1);   // out-proj rows -> caches

  for (int i = tid; i < 676; i += 256) rpe_lds[i] = rpe[i];

  // stage window tokens (rows >= 49 zero-padded), ds_store_b128
  const _Float16* xwg = xw + (size_t)win * (LW * C_);
  for (int i = tid; i < 64 * 16; i += 256) {
    int row = i >> 4;
    int cg  = (i & 15) << 3;
    uint4 val = make_uint4(0u, 0u, 0u, 0u);
    if (row < LW) val = *(const uint4*)(xwg + row * C_ + cg);
    *(uint4*)(xw_lds + row * LDP + cg) = val;
  }
  __syncthreads();

  // ---- QKV projection: (64 x 128) @ (128 x 384) -------------------------------
  const int mt = wv & 3;          // 16-row tile of tokens
  const int nside = wv >> 2;      // waves 0-3: cols 0..191, waves 4-7: 192..383
  v16h afq[4];
#pragma unroll
  for (int ks = 0; ks < 4; ++ks) afq[ks] = load_a_lds(xw_lds, mt * 16, ks * 32, LDP);

  const float qscale = 0.17677669529663687f;      // 32^-0.5
  for (int nt = 0; nt < 12; ++nt) {
    int n0 = nside * 192 + nt * 16;
    v8f acc = zero8();
#pragma unroll
    for (int ks = 0; ks < 4; ++ks)
      acc = wmma16(afq[ks], load_b_rowmajor(Wqkv, n0, ks * 32, C_), acc);
    int col = n0 + (lane & 15);
    float bias = bqkv[col];
    int rbase = mt * 16 + ((lane >> 4) << 3);
    if (col < 128) {               // Q (pre-scaled), row-major
#pragma unroll
      for (int i = 0; i < 8; ++i)
        q_lds[(rbase + i) * LDP + col] = (_Float16)((acc[i] + bias) * qscale);
    } else if (col < 256) {        // K, d-pair interleaved per head
      int dg = col - 128, hd = dg >> 5, d = dg & 31;
#pragma unroll
      for (int i = 0; i < 8; ++i)
        kp_lds[hd * 2048 + (d >> 1) * 128 + (rbase + i) * 2 + (d & 1)] =
            (_Float16)(acc[i] + bias);
    } else {                       // V, token-pair interleaved per head
      int dg = col - 256, hd = dg >> 5, d = dg & 31;
#pragma unroll
      for (int i = 0; i < 8; ++i) {
        int row = rbase + i;
        vp_lds[hd * 2048 + (row >> 1) * 64 + d * 2 + (row & 1)] =
            (_Float16)(acc[i] + bias);
      }
    }
  }
  __syncthreads();

  // ---- attention: 16 (head, row-tile) combos, 2 per wave ----------------------
#pragma unroll
  for (int cc = 0; cc < 2; ++cc) {
    int combo = wv + (cc << 3);
    int hd  = combo >> 2;
    int mt2 = combo & 3;
    // S = Q @ K^T  (16 x 64, single K-step of 32 = head dim)
    v16h aq = load_a_lds(q_lds, mt2 * 16, hd * 32, LDP);
    float s[4][8];
#pragma unroll
    for (int nt = 0; nt < 4; ++nt) {
      v8f acc = wmma16(aq, load_b_kp(kp_lds + hd * 2048, nt * 16), zero8());
#pragma unroll
      for (int i = 0; i < 8; ++i) s[nt][i] = acc[i];
    }
    int colb  = lane & 15;
    int rbase = mt2 * 16 + ((lane >> 4) << 3);
    // relative-position bias + key-padding mask
#pragma unroll
    for (int nt = 0; nt < 4; ++nt) {
      int kc = nt * 16 + colb;
#pragma unroll
      for (int i = 0; i < 8; ++i) {
        int r = rbase + i;
        if (kc >= LW) s[nt][i] = -1e30f;
        else if (r < LW) {
          int qi = r / WS_, qj = r - qi * WS_;
          int ki = kc / WS_, kj = kc - ki * WS_;
          s[nt][i] += rpe_lds[((qi - ki + 6) * 13 + (qj - kj + 6)) * NH_ + hd];
        }
      }
    }
    // row softmax: cross-lane masks 8/4/2/1 stay inside 16-lane halves,
    // which is exactly how C-layout rows are striped.
#pragma unroll
    for (int i = 0; i < 8; ++i) {
      float m = fmaxf(fmaxf(s[0][i], s[1][i]), fmaxf(s[2][i], s[3][i]));
#pragma unroll
      for (int off = 8; off >= 1; off >>= 1) m = fmaxf(m, __shfl_xor(m, off, 32));
      float sum = 0.f;
#pragma unroll
      for (int nt = 0; nt < 4; ++nt) {
        float p = __expf(s[nt][i] - m);
        s[nt][i] = p;
        sum += p;
      }
#pragma unroll
      for (int off = 8; off >= 1; off >>= 1) sum += __shfl_xor(sum, off, 32);
      float inv = 1.f / sum;
      int r = rbase + i;
#pragma unroll
      for (int nt = 0; nt < 4; ++nt)
        p_lds[hd * (64 * PSTR) + r * PSTR + nt * 16 + colb] =
            (_Float16)(s[nt][i] * inv);
    }
    // O = P @ V  (16 x 32, K = 64 tokens -> 2 K-steps)
#pragma unroll
    for (int nt2 = 0; nt2 < 2; ++nt2) {
      v8f acc = zero8();
#pragma unroll
      for (int ks = 0; ks < 2; ++ks)
        acc = wmma16(load_a_lds(p_lds + hd * (64 * PSTR), mt2 * 16, ks * 32, PSTR),
                     load_b_vp(vp_lds + hd * 2048, nt2 * 16, ks * 32), acc);
      int dcol = hd * 32 + nt2 * 16 + colb;
#pragma unroll
      for (int i = 0; i < 8; ++i)
        ao_lds[(rbase + i) * LDP + dcol] = (_Float16)acc[i];
    }
  }
  __syncthreads();

  // ---- output projection (64 x 128) @ (128 x 128) + un-window scatter ---------
  const int ww2 = win & 31;
  const int hh2 = (win >> 5) & 31;
  const int bb  = win >> 10;
  const int mt3 = wv & 3;
  const int nh2 = wv >> 2;
  v16h afo[4];
#pragma unroll
  for (int ks = 0; ks < 4; ++ks) afo[ks] = load_a_lds(ao_lds, mt3 * 16, ks * 32, LDP);
#pragma unroll
  for (int nt = 0; nt < 4; ++nt) {
    int n0 = nh2 * 64 + nt * 16;
    v8f acc = zero8();
#pragma unroll
    for (int ks = 0; ks < 4; ++ks)
      acc = wmma16(afo[ks], load_b_rowmajor(Wout, n0, ks * 32, C_), acc);
    int col = n0 + (lane & 15);
    float bias = bout[col];
    int rbase = mt3 * 16 + ((lane >> 4) << 3);
#pragma unroll
    for (int i = 0; i < 8; ++i) {
      int l = rbase + i;
      if (l < LW) {
        int li = l / WS_, lj = l - li * WS_;
        int n = (hh2 * WS_ + li) * WW_ + (ww2 * WS_ + lj);
        out[((size_t)bb * NPIX + n) * C_ + col] = acc[i] + bias;
      }
    }
  }
}

// ---------------- host entry --------------------------------------------------
extern "C" void kernel_launch(void* const* d_in, const int* in_sizes, int n_in,
                              void* d_out, int out_size, void* d_ws, size_t ws_size,
                              hipStream_t stream) {
  (void)in_sizes; (void)n_in; (void)out_size; (void)ws_size;
  const float* x     = (const float*)d_in[0];
  // d_in[1]=H, d_in[2]=W (compile-time constants here)
  const float* ca_w1 = (const float*)d_in[3];
  const float* ca_b1 = (const float*)d_in[4];
  const float* ca_w2 = (const float*)d_in[5];
  const float* ca_b2 = (const float*)d_in[6];
  const float* sa_w  = (const float*)d_in[7];
  const float* sa_b  = (const float*)d_in[8];
  const float* inw   = (const float*)d_in[9];
  const float* inb   = (const float*)d_in[10];
  const float* outw  = (const float*)d_in[11];
  const float* outb  = (const float*)d_in[12];
  const float* rpe   = (const float*)d_in[13];
  float* out = (float*)d_out;

  char* ws = (char*)d_ws;
  float*    pooled = (float*)(ws + OFF_POOLED);
  float*    sca    = (float*)(ws + OFF_SCA);
  float*    sain   = (float*)(ws + OFF_SAIN);
  float*    sasig  = (float*)(ws + OFF_SASIG);
  _Float16* winh   = (_Float16*)(ws + OFF_WIN);
  _Float16* wouth  = (_Float16*)(ws + OFF_WOUT);
  _Float16* xw     = (_Float16*)(ws + OFF_XW);

  k_zero <<<4, 256, 0, stream>>>(pooled, B_ * C_);
  k_pool <<<dim3(64, B_), C_, 0, stream>>>(x, pooled);
  k_ca   <<<B_, C_, 0, stream>>>(pooled, ca_w1, ca_b1, ca_w2, ca_b2, sca);
  k_stats<<<NTOK / 256, 256, 0, stream>>>(x, sca, sain);
  k_conv <<<NTOK / 256, 256, 0, stream>>>(sain, sa_w, sa_b, sasig);
  k_wprep<<<192, 256, 0, stream>>>(inw, outw, winh, wouth);
  k_pack <<<(NTOK * 32) / 256, 256, 0, stream>>>(x, sca, sasig, xw);

  (void)hipFuncSetAttribute((const void*)win_attn,
                            hipFuncAttributeMaxDynamicSharedMemorySize, SMEM_BYTES);
  win_attn<<<NWIN, 256, SMEM_BYTES, stream>>>(xw, winh, inb, wouth, outb, rpe, out);
}
